// DepthModule_85212151152983
// MI455X (gfx1250) — matrix operand order repaired and checked by reading, not compiled, not run on hardware
//
#include <hip/hip_runtime.h>

typedef __attribute__((ext_vector_type(16))) _Float16 v16h;
typedef __attribute__((ext_vector_type(8)))  float    v8f;

#define B_      32
#define CC      256
#define NH      8
#define HD      32
#define NTOK    49
#define M_ROWS  100352            // B*56*56 == B*64*49 == 3136*32
#define QK_SCALE 0.17677669529663687f
#define EPS_    1e-5f

#define WMMA_F16(a,b,c) \
  __builtin_amdgcn_wmma_f32_16x16x32_f16(false,(a),false,(b),(short)0,(c),false,false)

union Frag {
  v16h v;
  uint4 q[2];
  _Float16 h[16];
};

static __device__ __forceinline__ v8f vzero8() {
  v8f z = {0.f,0.f,0.f,0.f,0.f,0.f,0.f,0.f};
  return z;
}

// ---------------------------------------------------------------------------
// fp32 -> fp16 weight conversion
// ---------------------------------------------------------------------------
__global__ void cvt_f16_kernel(const float* __restrict__ in, _Float16* __restrict__ out, int n) {
  int i = blockIdx.x * 256 + threadIdx.x;
  if (i < n) out[i] = (_Float16)in[i];
}

// ---------------------------------------------------------------------------
// LayerNorm: mode 0 = LN1 + roll(-shift) + window partition -> f16
//            mode 1 = plain LN -> f16
//            mode 2 = plain LN -> f32 (final)
// one 256-thread block per output row
// ---------------------------------------------------------------------------
__global__ void ln_kernel(const float* __restrict__ x, const float* __restrict__ g,
                          const float* __restrict__ bta, void* out, int mode, int shift) {
  int m = blockIdx.x;
  int c = threadIdx.x;
  int lane = c & 31, wave = c >> 5;
  size_t src;
  if (mode == 0) {
    int win = m / NTOK, t = m - win * NTOK;
    int bb = win >> 6, widx = win & 63;
    int wh = widx >> 3, ww = widx & 7;
    int i = t / 7, j = t - i * 7;
    int hh = (wh * 7 + i + shift) % 56;
    int wp = (ww * 7 + j + shift) % 56;
    src = (size_t)bb * 3136 + hh * 56 + wp;
  } else {
    src = (size_t)m;
  }
  float v = x[src * CC + c];
  float s1 = v, s2 = v * v;
#pragma unroll
  for (int o = 1; o < 32; o <<= 1) { s1 += __shfl_xor(s1, o, 32); s2 += __shfl_xor(s2, o, 32); }
  __shared__ float r1[8], r2[8];
  if (lane == 0) { r1[wave] = s1; r2[wave] = s2; }
  __syncthreads();
  float sum = 0.f, sq = 0.f;
#pragma unroll
  for (int w = 0; w < 8; ++w) { sum += r1[w]; sq += r2[w]; }
  float mu  = sum * (1.f / CC);
  float var = sq * (1.f / CC) - mu * mu;
  float y = (v - mu) * rsqrtf(var + EPS_) * g[c] + bta[c];
  if (mode == 2) ((float*)out)[(size_t)m * CC + c] = y;
  else           ((_Float16*)out)[(size_t)m * CC + c] = (_Float16)y;
}

// ---------------------------------------------------------------------------
// 32x64 WMMA tile: acc[2][4] += A[m0:m0+32, :] @ W[n0:n0+64, :]^T
// A row-major [M,K] f16, W row-major [N,K] f16. K multiple of 32.
// Two A fragments share each B fragment: 8 WMMAs per 12 b128 loads per k-step,
// and each B fragment feeds two back-to-back WMMAs (second one wait-free).
// ---------------------------------------------------------------------------
template <int K>
static __device__ __forceinline__ void gemm_tile32(const _Float16* __restrict__ A,
                                                   const _Float16* __restrict__ W,
                                                   int m0, int n0, v8f acc[2][4]) {
  int lane = threadIdx.x & 31;
  int half = lane >> 4, l15 = lane & 15;
  const _Float16* arow0 = A + (size_t)(m0 + l15) * K + 8 * half;
  const _Float16* arow1 = arow0 + (size_t)16 * K;
  const _Float16* w0    = W + (size_t)(n0 + l15) * K + 8 * half;
#pragma unroll
  for (int k0 = 0; k0 < K; k0 += 32) {
    Frag a0, a1;
    a0.q[0] = *(const uint4*)(arow0 + k0);
    a0.q[1] = *(const uint4*)(arow0 + k0 + 16);
    a1.q[0] = *(const uint4*)(arow1 + k0);
    a1.q[1] = *(const uint4*)(arow1 + k0 + 16);
#pragma unroll
    for (int t = 0; t < 4; ++t) {
      const _Float16* wr = w0 + (size_t)(16 * t) * K + k0;
      Frag b;
      b.q[0] = *(const uint4*)(wr);
      b.q[1] = *(const uint4*)(wr + 16);
      acc[0][t] = WMMA_F16(a0.v, b.v, acc[0][t]);
      acc[1][t] = WMMA_F16(a1.v, b.v, acc[1][t]);
    }
  }
}

// ---------------------------------------------------------------------------
// QKV GEMM: [M,256] @ [768,256]^T + bias -> f16 [M,768]
// ---------------------------------------------------------------------------
__global__ void gemm_qkv_kernel(const _Float16* __restrict__ A, const _Float16* __restrict__ W,
                                const float* __restrict__ bias, _Float16* __restrict__ out) {
  const int NT = 768 / 64;
  int wid = blockIdx.x * 8 + (threadIdx.x >> 5);
  if (wid >= (M_ROWS / 32) * NT) return;
  int mt = wid / NT, nt = wid % NT;
  int m0 = mt * 32, n0 = nt * 64;
  v8f acc[2][4];
#pragma unroll
  for (int rr = 0; rr < 2; ++rr)
#pragma unroll
    for (int t = 0; t < 4; ++t) acc[rr][t] = vzero8();
  gemm_tile32<256>(A, W, m0, n0, acc);
  int lane = threadIdx.x & 31, half = lane >> 4, l15 = lane & 15;
#pragma unroll
  for (int t = 0; t < 4; ++t) {
    int n = n0 + 16 * t + l15;
    float bv = bias[n];
#pragma unroll
    for (int rr = 0; rr < 2; ++rr)
#pragma unroll
      for (int r = 0; r < 8; ++r) {
        int m = m0 + rr * 16 + r + 8 * half;
        out[(size_t)m * 768 + n] = (_Float16)(acc[rr][t][r] + bv);
      }
  }
}

// ---------------------------------------------------------------------------
// Windowed attention, one block (128 thr, 4 waves) per (window, head).
// qkv layout [B*NW*49, 768]: q @ +0, k @ +256, v @ +512 (head h at h*32).
// ---------------------------------------------------------------------------
__global__ void attn_kernel(const _Float16* __restrict__ qkv, const float* __restrict__ rpb,
                            _Float16* __restrict__ out, int shift) {
  int win  = blockIdx.x;          // 0..2047
  int hidx = blockIdx.y;          // 0..7
  int wave = threadIdx.x >> 5;
  int lane = threadIdx.x & 31, half = lane >> 4, l15 = lane & 15;
  int m0 = wave * 16;
  size_t wbase = (size_t)win * NTOK;
  uint4 z4; z4.x = z4.y = z4.z = z4.w = 0u;

  // A fragment = Q rows (zero-padded past 49)
  Frag a; a.q[0] = z4; a.q[1] = z4;
  int qt = m0 + l15;
  if (qt < NTOK) {
    const _Float16* p = qkv + (wbase + qt) * 768 + hidx * HD;
    a.q[0] = *(const uint4*)(p + 8 * half);
    a.q[1] = *(const uint4*)(p + 16 + 8 * half);
  }

  // S = Q @ K^T  (K-dim = HD = 32, one WMMA per 16x16 tile)
  v8f s[4];
#pragma unroll
  for (int nt = 0; nt < 4; ++nt) s[nt] = vzero8();
#pragma unroll
  for (int nt = 0; nt < 4; ++nt) {
    int kt = nt * 16 + l15;
    Frag b; b.q[0] = z4; b.q[1] = z4;
    if (kt < NTOK) {
      const _Float16* p = qkv + (wbase + kt) * 768 + 256 + hidx * HD + 8 * half;
      b.q[0] = *(const uint4*)(p);
      b.q[1] = *(const uint4*)(p + 16);
    }
    s[nt] = WMMA_F16(a.v, b.v, s[nt]);
  }

  // scale + relative position bias + shift mask + column padding
  int widx = win & 63;
  int wh = widx >> 3, ww = widx & 7;
#pragma unroll
  for (int r = 0; r < 8; ++r) {
    int qrow = m0 + r + 8 * half;
    int qi = qrow / 7, qj = qrow - qi * 7;       // only meaningful if qrow<49
    int qh = wh * 7 + qi, qw = ww * 7 + qj;
    int rqh = qh < 49 ? 0 : (qh < 53 ? 1 : 2);
    int rqw = qw < 49 ? 0 : (qw < 53 ? 1 : 2);
#pragma unroll
    for (int nt = 0; nt < 4; ++nt) {
      int kc = nt * 16 + l15;
      float v = s[nt][r] * QK_SCALE;
      if (kc >= NTOK) {
        v = -1e30f;
      } else if (qrow < NTOK) {
        int ki = kc / 7, kj = kc - ki * 7;
        int rel = (qi - ki + 6) * 13 + (qj - kj + 6);
        v += rpb[rel * NH + hidx];
        if (shift) {
          int kh = wh * 7 + ki, kw = ww * 7 + kj;
          int rkh = kh < 49 ? 0 : (kh < 53 ? 1 : 2);
          int rkw = kw < 49 ? 0 : (kw < 53 ? 1 : 2);
          if (rkh != rqh || rkw != rqw) v += -100.0f;
        }
      }
      s[nt][r] = v;
    }
  }

  // softmax across the 64 (padded) columns: in-register + 16-lane xor reduce
#pragma unroll
  for (int r = 0; r < 8; ++r) {
    float mx = fmaxf(fmaxf(s[0][r], s[1][r]), fmaxf(s[2][r], s[3][r]));
#pragma unroll
    for (int o = 1; o < 16; o <<= 1) mx = fmaxf(mx, __shfl_xor(mx, o, 32));
    float sum = 0.f;
#pragma unroll
    for (int nt = 0; nt < 4; ++nt) { float e = __expf(s[nt][r] - mx); s[nt][r] = e; sum += e; }
#pragma unroll
    for (int o = 1; o < 16; o <<= 1) sum += __shfl_xor(sum, o, 32);
    float inv = 1.f / sum;
#pragma unroll
    for (int nt = 0; nt < 4; ++nt) s[nt][r] *= inv;
  }

  // stage P (f16) in wave-private LDS, re-read as A fragments
  __shared__ _Float16 lds_p[4][16][64];
#pragma unroll
  for (int r = 0; r < 8; ++r)
#pragma unroll
    for (int nt = 0; nt < 4; ++nt)
      lds_p[wave][r + 8 * half][nt * 16 + l15] = (_Float16)s[nt][r];
  __syncthreads();

  // O = P @ V   (K-dim 64 padded -> 2 WMMA steps, 2 N-tiles for HD=32)
  v8f o2[2];
  o2[0] = vzero8(); o2[1] = vzero8();
#pragma unroll
  for (int kk = 0; kk < 2; ++kk) {
    Frag pa;
    const _Float16* pr = &lds_p[wave][l15][kk * 32 + 8 * half];
    pa.q[0] = *(const uint4*)(pr);
    pa.q[1] = *(const uint4*)(pr + 16);
#pragma unroll
    for (int nt = 0; nt < 2; ++nt) {
      Frag vb;
#pragma unroll
      for (int j = 0; j < 16; ++j) {
        int key = kk * 32 + (j < 8 ? 8 * half + j : 16 + 8 * half + (j - 8));
        vb.h[j] = (key < NTOK) ? qkv[(wbase + key) * 768 + 512 + hidx * HD + nt * 16 + l15]
                               : (_Float16)0.0f;
      }
      o2[nt] = WMMA_F16(pa.v, vb.v, o2[nt]);
    }
  }

#pragma unroll
  for (int nt = 0; nt < 2; ++nt)
#pragma unroll
    for (int r = 0; r < 8; ++r) {
      int qrow = m0 + r + 8 * half;
      if (qrow < NTOK)
        out[(wbase + qrow) * CC + hidx * HD + nt * 16 + l15] = (_Float16)o2[nt][r];
    }
}

// ---------------------------------------------------------------------------
// proj GEMM + window reverse + roll(+shift) + residual -> f32
// ---------------------------------------------------------------------------
__global__ void gemm_proj_kernel(const _Float16* __restrict__ A, const _Float16* __restrict__ W,
                                 const float* __restrict__ bias, const float* __restrict__ xin,
                                 float* __restrict__ xout, int shift) {
  const int NT = 4;
  int wid = blockIdx.x * 8 + (threadIdx.x >> 5);
  if (wid >= (M_ROWS / 32) * NT) return;
  int mt = wid / NT, nt = wid % NT;
  int m0 = mt * 32, n0 = nt * 64;
  v8f acc[2][4];
#pragma unroll
  for (int rr = 0; rr < 2; ++rr)
#pragma unroll
    for (int t = 0; t < 4; ++t) acc[rr][t] = vzero8();
  gemm_tile32<256>(A, W, m0, n0, acc);
  int lane = threadIdx.x & 31, half = lane >> 4, l15 = lane & 15;
#pragma unroll
  for (int rr = 0; rr < 2; ++rr)
#pragma unroll
    for (int r = 0; r < 8; ++r) {
      int m = m0 + rr * 16 + r + 8 * half;
      int win = m / NTOK, t = m - win * NTOK;
      int bb = win >> 6, widx = win & 63;
      int wh = widx >> 3, ww = widx & 7;
      int i = t / 7, j = t - i * 7;
      int hh = (wh * 7 + i + shift) % 56;
      int wp = (ww * 7 + j + shift) % 56;
      size_t row = ((size_t)bb * 3136 + hh * 56 + wp) * CC;
#pragma unroll
      for (int tt = 0; tt < 4; ++tt) {
        int n = n0 + 16 * tt + l15;
        xout[row + n] = xin[row + n] + acc[rr][tt][r] + bias[n];
      }
    }
}

// ---------------------------------------------------------------------------
// fc1 GEMM + exact GELU -> f16 [M,1024]
// ---------------------------------------------------------------------------
__global__ void gemm_fc1_kernel(const _Float16* __restrict__ A, const _Float16* __restrict__ W,
                                const float* __restrict__ bias, _Float16* __restrict__ out) {
  const int NT = 1024 / 64;
  int wid = blockIdx.x * 8 + (threadIdx.x >> 5);
  if (wid >= (M_ROWS / 32) * NT) return;
  int mt = wid / NT, nt = wid % NT;
  int m0 = mt * 32, n0 = nt * 64;
  v8f acc[2][4];
#pragma unroll
  for (int rr = 0; rr < 2; ++rr)
#pragma unroll
    for (int t = 0; t < 4; ++t) acc[rr][t] = vzero8();
  gemm_tile32<256>(A, W, m0, n0, acc);
  int lane = threadIdx.x & 31, half = lane >> 4, l15 = lane & 15;
#pragma unroll
  for (int t = 0; t < 4; ++t) {
    int n = n0 + 16 * t + l15;
    float bv = bias[n];
#pragma unroll
    for (int rr = 0; rr < 2; ++rr)
#pragma unroll
      for (int r = 0; r < 8; ++r) {
        int m = m0 + rr * 16 + r + 8 * half;
        float v = acc[rr][t][r] + bv;
        float g = 0.5f * v * (1.0f + erff(v * 0.70710678118654752f));
        out[(size_t)m * 1024 + n] = (_Float16)g;
      }
  }
}

// ---------------------------------------------------------------------------
// fc2 GEMM (K=1024) + residual -> f32 [M,256]
// ---------------------------------------------------------------------------
__global__ void gemm_fc2_kernel(const _Float16* __restrict__ A, const _Float16* __restrict__ W,
                                const float* __restrict__ bias, const float* __restrict__ xmid,
                                float* __restrict__ xout) {
  const int NT = 4;
  int wid = blockIdx.x * 8 + (threadIdx.x >> 5);
  if (wid >= (M_ROWS / 32) * NT) return;
  int mt = wid / NT, nt = wid % NT;
  int m0 = mt * 32, n0 = nt * 64;
  v8f acc[2][4];
#pragma unroll
  for (int rr = 0; rr < 2; ++rr)
#pragma unroll
    for (int t = 0; t < 4; ++t) acc[rr][t] = vzero8();
  gemm_tile32<1024>(A, W, m0, n0, acc);
  int lane = threadIdx.x & 31, half = lane >> 4, l15 = lane & 15;
#pragma unroll
  for (int t = 0; t < 4; ++t) {
    int n = n0 + 16 * t + l15;
    float bv = bias[n];
#pragma unroll
    for (int rr = 0; rr < 2; ++rr)
#pragma unroll
      for (int r = 0; r < 8; ++r) {
        size_t row = (size_t)(m0 + rr * 16 + r + 8 * half) * CC;
        xout[row + n] = xmid[row + n] + acc[rr][t][r] + bv;
      }
  }
}

// ---------------------------------------------------------------------------
extern "C" void kernel_launch(void* const* d_in, const int* in_sizes, int n_in,
                              void* d_out, int out_size, void* d_ws, size_t ws_size,
                              hipStream_t stream) {
  const float* x      = (const float*)d_in[0];
  const float* qkv_w  = (const float*)d_in[1];
  const float* qkv_b  = (const float*)d_in[2];
  const float* proj_w = (const float*)d_in[3];
  const float* proj_b = (const float*)d_in[4];
  const float* rpb    = (const float*)d_in[5];
  const float* n1g    = (const float*)d_in[6];
  const float* n1b    = (const float*)d_in[7];
  const float* n2g    = (const float*)d_in[8];
  const float* n2b    = (const float*)d_in[9];
  const float* f1w    = (const float*)d_in[10];
  const float* f1b    = (const float*)d_in[11];
  const float* f2w    = (const float*)d_in[12];
  const float* f2b    = (const float*)d_in[13];
  const float* nfg    = (const float*)d_in[14];
  const float* nfb    = (const float*)d_in[15];

  char* ws = (char*)d_ws;
  size_t off = 0;
  float*    X0    = (float*)(ws + off);     off += (size_t)M_ROWS * CC * 4;
  float*    X1    = (float*)(ws + off);     off += (size_t)M_ROWS * CC * 4;
  _Float16* bufA  = (_Float16*)(ws + off);  off += (size_t)M_ROWS * 1024 * 2;   // qkv / mlp hidden
  _Float16* bufB  = (_Float16*)(ws + off);  off += (size_t)M_ROWS * CC * 2;     // ln out / attn out
  _Float16* wqkv  = (_Float16*)(ws + off);  off += (size_t)3 * 768 * 256 * 2;
  _Float16* wproj = (_Float16*)(ws + off);  off += (size_t)3 * 256 * 256 * 2;
  _Float16* wfc1  = (_Float16*)(ws + off);  off += (size_t)3 * 1024 * 256 * 2;
  _Float16* wfc2  = (_Float16*)(ws + off);  off += (size_t)3 * 256 * 1024 * 2;

  cvt_f16_kernel<<<(3 * 768 * 256 + 255) / 256, 256, 0, stream>>>(qkv_w, wqkv, 3 * 768 * 256);
  cvt_f16_kernel<<<(3 * 256 * 256 + 255) / 256, 256, 0, stream>>>(proj_w, wproj, 3 * 256 * 256);
  cvt_f16_kernel<<<(3 * 1024 * 256 + 255) / 256, 256, 0, stream>>>(f1w, wfc1, 3 * 1024 * 256);
  cvt_f16_kernel<<<(3 * 256 * 1024 + 255) / 256, 256, 0, stream>>>(f2w, wfc2, 3 * 256 * 1024);

  const float* xin = x;
  for (int d = 0; d < 3; ++d) {
    int shift = (d & 1) ? 3 : 0;
    // LN1 + shift + window partition
    ln_kernel<<<M_ROWS, 256, 0, stream>>>(xin, n1g + d * 256, n1b + d * 256, bufB, 0, shift);
    // qkv: 3136*12 tiles / 8 waves
    gemm_qkv_kernel<<<4704, 256, 0, stream>>>(bufB, wqkv + (size_t)d * 768 * 256,
                                              qkv_b + d * 768, bufA);
    // attention
    dim3 ag(2048, 8);
    attn_kernel<<<ag, 128, 0, stream>>>(bufA, rpb + (size_t)d * 169 * NH, bufB, shift);
    // proj + window reverse + residual: 3136*4 tiles / 8
    gemm_proj_kernel<<<1568, 256, 0, stream>>>(bufB, wproj + (size_t)d * 256 * 256,
                                               proj_b + d * 256, xin, X1, shift);
    // LN2
    ln_kernel<<<M_ROWS, 256, 0, stream>>>(X1, n2g + d * 256, n2b + d * 256, bufB, 1, 0);
    // fc1 + gelu: 3136*16 tiles / 8
    gemm_fc1_kernel<<<6272, 256, 0, stream>>>(bufB, wfc1 + (size_t)d * 1024 * 256,
                                              f1b + d * 1024, bufA);
    // fc2 + residual: 3136*4 tiles / 8
    gemm_fc2_kernel<<<1568, 256, 0, stream>>>(bufA, wfc2 + (size_t)d * 256 * 1024,
                                              f2b + d * 256, X1, X0);
    xin = X0;
  }
  ln_kernel<<<M_ROWS, 256, 0, stream>>>(X0, nfg, nfb, d_out, 2, 0);
}
